// Attention_39926015984133
// MI455X (gfx1250) — matrix operand order repaired and checked by reading, not compiled, not run on hardware
//
#include <hip/hip_runtime.h>

// ---------------------------------------------------------------------------
// MI455X (gfx1250) attention block: y = ((softmax((xWq^T)(xWk^T)^T / D)) (xWv^T)) Wp^T
// Compute-bound (~0.41 TFLOP vs ~150MB unavoidable traffic @ 23.3TB/s) ->
// bf16 WMMA everywhere, fp32 accumulate, online softmax in fp32.
// Assumes ws_size >= ~176 MB.
// ---------------------------------------------------------------------------

typedef __attribute__((ext_vector_type(16))) __bf16 v16bf;
typedef __attribute__((ext_vector_type(8)))  float  v8f;
typedef __attribute__((ext_vector_type(4)))  unsigned int v4u;
typedef __attribute__((ext_vector_type(4)))  float  v4f;

#define BATCH 4
#define SLEN  4096
#define DIMD  1024
#define MTOT  (BATCH * SLEN)   // 16384 rows for the projections

union Frag {
    v16bf v;
    v4u   q[2];
};

__device__ __forceinline__ unsigned short f2bfbits(float f) {
    unsigned u = __builtin_bit_cast(unsigned, f);
    return (unsigned short)((u + 0x7FFFu + ((u >> 16) & 1u)) >> 16);
}

__device__ __forceinline__ v8f zero8() {
    v8f z;
#pragma unroll
    for (int i = 0; i < 8; ++i) z[i] = 0.0f;
    return z;
}

// A-operand (16 x 32 bf16, row-major source, leading dim ld), k order used here (same as B):
// lane(r = lane&15, hi = lane>>4): elems 0..15 -> K = kb + hi*16 + e (contiguous)
__device__ __forceinline__ Frag load_frag_a(const unsigned short* base, int ld,
                                            int row0, int kb, int lane) {
    const int r = lane & 15, hi = lane >> 4;
    const unsigned short* p = base + (size_t)(row0 + r) * ld + kb + hi * 16;
    Frag f;
    f.q[0] = *(const v4u*)p;
    f.q[1] = *(const v4u*)(p + 8);
    return f;
}

// B-operand (32 x 16 bf16). Column n of B == row (col0+n) of the row-major
// source (W stored [E][D], K stored [S][D], Vt stored [D][S]).
// lane(n = lane&15, hi = lane>>4): elems 0..15 -> K = kb + hi*16 + e (contiguous)
__device__ __forceinline__ Frag load_frag_b(const unsigned short* base, int ld,
                                            int col0, int kb, int lane) {
    const int n = lane & 15, hi = lane >> 4;
    const unsigned short* p = base + (size_t)(col0 + n) * ld + kb + hi * 16;
    Frag f;
    f.q[0] = *(const v4u*)p;
    f.q[1] = *(const v4u*)(p + 8);
    return f;
}

__device__ __forceinline__ v8f wmma_bf16(const Frag& a, const Frag& b, v8f c) {
    return __builtin_amdgcn_wmma_f32_16x16x32_bf16(
        /*neg_a=*/false, a.v, /*neg_b=*/false, b.v,
        /*c_mod=*/(short)0, c, /*reuse_a=*/false, /*reuse_b=*/false);
}

// ---------------------------------------------------------------------------
// fp32 -> bf16 convert, 8 elements / thread (b128 in, b128 out)
// ---------------------------------------------------------------------------
__global__ __launch_bounds__(256) void cvt_f32_bf16(const float* __restrict__ in,
                                                    unsigned short* __restrict__ out,
                                                    int n) {
    int i = (blockIdx.x * 256 + threadIdx.x) * 8;
    if (i + 8 > n) return;
    v4f f0 = *(const v4f*)(in + i);
    v4f f1 = *(const v4f*)(in + i + 4);
    v4u o;
    o.x = (unsigned)f2bfbits(f0.x) | ((unsigned)f2bfbits(f0.y) << 16);
    o.y = (unsigned)f2bfbits(f0.z) | ((unsigned)f2bfbits(f0.w) << 16);
    o.z = (unsigned)f2bfbits(f1.x) | ((unsigned)f2bfbits(f1.y) << 16);
    o.w = (unsigned)f2bfbits(f1.z) | ((unsigned)f2bfbits(f1.w) << 16);
    *(v4u*)(out + i) = o;
}

// ---------------------------------------------------------------------------
// C[M x N] = A[M x K] * B[N x K]^T   (both bf16, fp32 accumulate)
// Block: 256 threads = 8 waves; block tile 128x128; wave tile 32x64.
// B block tile (128 cols x 32 k = 8KB) is staged in LDS once per K-step with
// gfx1250 GLOBAL_LOAD_ASYNC_TO_LDS_B128 (ASYNCcnt), shared by all 8 waves.
// mode 0: store bf16 row-major          (Q, K projections)
// mode 1: store bf16 transposed per 4096-row batch: out[b][n][s]  (V -> Vt)
// mode 2: store fp32 row-major          (final projection -> d_out)
// ---------------------------------------------------------------------------
__global__ __launch_bounds__(256) void gemm_bf16_wmma(
    const unsigned short* __restrict__ A, const unsigned short* __restrict__ B,
    void* __restrict__ C, int M, int N, int K, int mode) {
    __shared__ __attribute__((aligned(16))) unsigned short bs[128][32];  // [col][k]

    const int lane = threadIdx.x & 31;
    const int w    = threadIdx.x >> 5;
    const int wm   = w >> 1, wn = w & 1;
    const int m0   = blockIdx.y * 128 + wm * 32;
    const int nblk = blockIdx.x * 128;
    const int n0l  = wn * 64;              // wave's column base within block tile
    // cooperative B-stage assignment: thread copies 8 bf16 (one b128) into its 16-elem slot
    const int ccol = threadIdx.x >> 1;     // 0..127
    const int chalf = (threadIdx.x & 1) * 16;

    v8f acc[2][4];
#pragma unroll
    for (int i = 0; i < 2; ++i)
#pragma unroll
        for (int j = 0; j < 4; ++j) acc[i][j] = zero8();

    for (int kb = 0; kb < K; kb += 32) {
        if (kb) __syncthreads();  // previous tile fully consumed
        {
            const unsigned short* gsrc = B + (size_t)(nblk + ccol) * K + kb + chalf;
            unsigned lds_off = (unsigned)(size_t)(void*)&bs[ccol][chalf];
            // async DMA: 16 bytes global -> LDS per lane, tracked by ASYNCcnt
            asm volatile("global_load_async_to_lds_b128 %0, %1, off"
                         :: "v"(lds_off), "v"(gsrc) : "memory");
            asm volatile("s_wait_asynccnt 0x0" ::: "memory");
        }
        __syncthreads();

        Frag a0 = load_frag_a(A, K, m0, kb, lane);
        Frag a1 = load_frag_a(A, K, m0 + 16, kb, lane);
        if (kb + 32 < K) {  // gfx1250 global_prefetch_b8 of next A tile
            __builtin_prefetch(A + (size_t)(m0 + (lane & 15)) * K + kb + 32, 0, 1);
        }
        Frag bf[4];
#pragma unroll
        for (int j = 0; j < 4; ++j) {  // B frag from LDS: 16 contiguous bf16
            const unsigned short* p = &bs[n0l + j * 16 + (lane & 15)][(lane >> 4) * 16];
            bf[j].q[0] = *(const v4u*)p;
            bf[j].q[1] = *(const v4u*)(p + 8);
        }
#pragma unroll
        for (int j = 0; j < 4; ++j) {
            acc[0][j] = wmma_bf16(a0, bf[j], acc[0][j]);
            acc[1][j] = wmma_bf16(a1, bf[j], acc[1][j]);
        }
    }

    // C/D layout: VGPR e, lanes 0-15 -> (M=e, N=lane); lanes 16-31 -> (M=8+e, N=lane-16)
    const int hi = lane >> 4, nl = lane & 15;
#pragma unroll
    for (int i = 0; i < 2; ++i)
#pragma unroll
        for (int j = 0; j < 4; ++j)
#pragma unroll
            for (int e = 0; e < 8; ++e) {
                const int m = m0 + i * 16 + hi * 8 + e;
                const int n = nblk + n0l + j * 16 + nl;
                const float val = acc[i][j][e];
                if (mode == 0) {
                    ((unsigned short*)C)[(size_t)m * N + n] = f2bfbits(val);
                } else if (mode == 1) {
                    const int b = m >> 12, s = m & (SLEN - 1);
                    ((unsigned short*)C)[((size_t)b * N + n) * SLEN + s] = f2bfbits(val);
                } else {
                    ((float*)C)[(size_t)m * N + n] = val;
                }
            }
}

// ---------------------------------------------------------------------------
// Flash attention: one block per (32-query tile, batch). 8 waves.
// Per 128-key chunk:
//   1) wave w computes a 32x16 score strip over K=D (64 WMMA, B frag reused x2)
//   2) wave-parallel online softmax (4 rows/wave, 8-lane groups)
//   3) each wave updates its 128 output columns: O += P(32x128) x V(128x128)
// ---------------------------------------------------------------------------
__global__ __launch_bounds__(256) void attn_flash(
    const unsigned short* __restrict__ Qg, const unsigned short* __restrict__ Kg,
    const unsigned short* __restrict__ Vt, unsigned short* __restrict__ Og) {
    __shared__ float s_scores[32][128];                                   // 16 KB
    __shared__ __attribute__((aligned(16))) unsigned short s_p[32][128];  //  8 KB
    __shared__ float s_m[32], s_l[32], s_scale[32];

    const int lane = threadIdx.x & 31;
    const int w    = threadIdx.x >> 5;
    const int hi   = lane >> 4, nl = lane & 15;
    const int b    = blockIdx.y;
    const int qrow0 = blockIdx.x * 32;
    const int d0   = w * 128;  // this wave's 128 output columns

    const unsigned short* Qb = Qg + ((size_t)b * SLEN + qrow0) * DIMD;
    const unsigned short* Kb = Kg + (size_t)b * SLEN * DIMD;
    const unsigned short* Vb = Vt + (size_t)b * DIMD * SLEN;

    if (threadIdx.x < 32) { s_m[threadIdx.x] = -3.0e38f; s_l[threadIdx.x] = 0.0f; }

    v8f oacc[2][8];
#pragma unroll
    for (int i = 0; i < 2; ++i)
#pragma unroll
        for (int j = 0; j < 8; ++j) oacc[i][j] = zero8();
    __syncthreads();

    for (int kc = 0; kc < SLEN; kc += 128) {
        // ---- 1) scores: wave w owns key columns [kc + 16w, kc + 16w + 16) ----
        v8f s0 = zero8(), s1 = zero8();
        const int kt = kc + w * 16;
        for (int kb = 0; kb < DIMD; kb += 32) {
            Frag a0  = load_frag_a(Qb, DIMD, 0, kb, lane);
            Frag a1  = load_frag_a(Qb, DIMD, 16, kb, lane);
            Frag bfr = load_frag_b(Kb, DIMD, kt, kb, lane);  // B col n = K row kt+n
            s0 = wmma_bf16(a0, bfr, s0);
            s1 = wmma_bf16(a1, bfr, s1);
        }
#pragma unroll
        for (int e = 0; e < 8; ++e) {
            s_scores[hi * 8 + e][w * 16 + nl]      = s0[e] * (1.0f / (float)DIMD);
            s_scores[16 + hi * 8 + e][w * 16 + nl] = s1[e] * (1.0f / (float)DIMD);
        }
        __syncthreads();

        // ---- 2) online softmax: wave w handles rows 4w..4w+3 (8 lanes/row) ----
        {
            const int row = 4 * w + (lane >> 3);
            const int c0  = lane & 7;
            float vv[16];
            float mloc = -3.0e38f;
#pragma unroll
            for (int it = 0; it < 16; ++it) {
                vv[it] = s_scores[row][c0 + 8 * it];
                mloc = fmaxf(mloc, vv[it]);
            }
            for (int off = 1; off < 8; off <<= 1)
                mloc = fmaxf(mloc, __shfl_xor(mloc, off, 32));
            const float mold = s_m[row];
            const float mnew = fmaxf(mold, mloc);
            float sum = 0.0f;
#pragma unroll
            for (int it = 0; it < 16; ++it) {
                const float p = __expf(vv[it] - mnew);
                sum += p;
                s_p[row][c0 + 8 * it] = f2bfbits(p);
            }
            for (int off = 1; off < 8; off <<= 1) sum += __shfl_xor(sum, off, 32);
            if (c0 == 0) {
                const float fac = __expf(mold - mnew);
                s_scale[row] = fac;
                s_m[row]     = mnew;
                s_l[row]     = s_l[row] * fac + sum;
            }
        }
        __syncthreads();

        // ---- 3) rescale accumulators, then O += P * V over this chunk ----
        float fac[2][8];
#pragma unroll
        for (int i = 0; i < 2; ++i)
#pragma unroll
            for (int e = 0; e < 8; ++e) fac[i][e] = s_scale[i * 16 + hi * 8 + e];
#pragma unroll
        for (int i = 0; i < 2; ++i)
#pragma unroll
            for (int j = 0; j < 8; ++j)
#pragma unroll
                for (int e = 0; e < 8; ++e) oacc[i][j][e] *= fac[i][e];

        for (int kb4 = 0; kb4 < 128; kb4 += 32) {
            Frag p0, p1;  // A-operand from LDS (P tile), contiguous k order (matches vb)
            {
                const int r = lane & 15;
                p0.q[0] = *(const v4u*)&s_p[r][kb4 + hi * 16];
                p0.q[1] = *(const v4u*)&s_p[r][kb4 + hi * 16 + 8];
                p1.q[0] = *(const v4u*)&s_p[16 + r][kb4 + hi * 16];
                p1.q[1] = *(const v4u*)&s_p[16 + r][kb4 + hi * 16 + 8];
            }
#pragma unroll
            for (int j = 0; j < 8; ++j) {
                Frag vb;  // B col n = Vt row (d0 + 16j + n): contiguous keys
                const unsigned short* p =
                    Vb + (size_t)(d0 + j * 16 + nl) * SLEN + kc + kb4 + hi * 16;
                vb.q[0] = *(const v4u*)p;
                vb.q[1] = *(const v4u*)(p + 8);
                oacc[0][j] = wmma_bf16(p0, vb, oacc[0][j]);
                oacc[1][j] = wmma_bf16(p1, vb, oacc[1][j]);
            }
        }
        __syncthreads();
    }

    // ---- epilogue: normalize by running sum, store bf16 ----
    float invl[2][8];
#pragma unroll
    for (int i = 0; i < 2; ++i)
#pragma unroll
        for (int e = 0; e < 8; ++e) invl[i][e] = 1.0f / s_l[i * 16 + hi * 8 + e];
#pragma unroll
    for (int i = 0; i < 2; ++i)
#pragma unroll
        for (int j = 0; j < 8; ++j)
#pragma unroll
            for (int e = 0; e < 8; ++e) {
                const int m = qrow0 + i * 16 + hi * 8 + e;
                const int d = d0 + j * 16 + nl;
                Og[((size_t)b * SLEN + m) * DIMD + d] =
                    f2bfbits(oacc[i][j][e] * invl[i][e]);
            }
}

// ---------------------------------------------------------------------------
extern "C" void kernel_launch(void* const* d_in, const int* in_sizes, int n_in,
                              void* d_out, int out_size, void* d_ws, size_t ws_size,
                              hipStream_t stream) {
    (void)in_sizes; (void)n_in; (void)out_size; (void)ws_size;
    const float* x  = (const float*)d_in[0];
    const float* Wq = (const float*)d_in[1];
    const float* Wk = (const float*)d_in[2];
    const float* Wv = (const float*)d_in[3];
    const float* Wp = (const float*)d_in[4];

    char* ws = (char*)d_ws;
    const size_t XB_BYTES = (size_t)MTOT * DIMD * 2;  // 32 MB
    const size_t W_BYTES  = (size_t)DIMD * DIMD * 2;  //  2 MB
    size_t off = 0;
    unsigned short* xb  = (unsigned short*)(ws + off); off += XB_BYTES;
    unsigned short* wqb = (unsigned short*)(ws + off); off += W_BYTES;
    unsigned short* wkb = (unsigned short*)(ws + off); off += W_BYTES;
    unsigned short* wvb = (unsigned short*)(ws + off); off += W_BYTES;
    unsigned short* wpb = (unsigned short*)(ws + off); off += W_BYTES;
    unsigned short* qb  = (unsigned short*)(ws + off); off += XB_BYTES;
    unsigned short* kb  = (unsigned short*)(ws + off); off += XB_BYTES;
    unsigned short* vt  = (unsigned short*)(ws + off); off += XB_BYTES;
    unsigned short* ob  = (unsigned short*)(ws + off); off += XB_BYTES;

    const int nx = MTOT * DIMD;        // 16,777,216
    const int nw = DIMD * DIMD;        //  1,048,576
    cvt_f32_bf16<<<nx / 2048, 256, 0, stream>>>(x,  xb,  nx);
    cvt_f32_bf16<<<nw / 2048, 256, 0, stream>>>(Wq, wqb, nw);
    cvt_f32_bf16<<<nw / 2048, 256, 0, stream>>>(Wk, wkb, nw);
    cvt_f32_bf16<<<nw / 2048, 256, 0, stream>>>(Wv, wvb, nw);
    cvt_f32_bf16<<<nw / 2048, 256, 0, stream>>>(Wp, wpb, nw);

    dim3 gemm_grid(DIMD / 128, MTOT / 128);  // (8, 128)
    gemm_bf16_wmma<<<gemm_grid, 256, 0, stream>>>(xb, wqb, qb, MTOT, DIMD, DIMD, 0);
    gemm_bf16_wmma<<<gemm_grid, 256, 0, stream>>>(xb, wkb, kb, MTOT, DIMD, DIMD, 0);
    gemm_bf16_wmma<<<gemm_grid, 256, 0, stream>>>(xb, wvb, vt, MTOT, DIMD, DIMD, 1);

    dim3 attn_grid(SLEN / 32, BATCH);        // (128, 4)
    attn_flash<<<attn_grid, 256, 0, stream>>>(qb, kb, vt, ob);

    gemm_bf16_wmma<<<gemm_grid, 256, 0, stream>>>(ob, wpb, d_out, MTOT, DIMD, DIMD, 2);
}
